// MemoLayer_2491081032430
// MI455X (gfx1250) — matrix operand order repaired and checked by reading, not compiled, not run on hardware
//
#include <hip/hip_runtime.h>
#include <hip/hip_bf16.h>
#include <math.h>

// ---------------- CDNA5 (gfx1250) wave32 WMMA types ----------------
typedef __attribute__((ext_vector_type(16))) __bf16 bf16x16;
typedef __attribute__((ext_vector_type(4)))  __bf16 bf16x4;
typedef __attribute__((ext_vector_type(2)))  __bf16 bf16x2;
typedef __attribute__((ext_vector_type(8)))  float  f32x8;

#define WMMA_BF16(a, b, c) \
  __builtin_amdgcn_wmma_f32_16x16x32_bf16(false, (a), false, (b), (short)0, (c), false, false)

// ---------------- problem constants ----------------
#define Bz   2
#define Sq   1024
#define Dm   1024
#define Hh   16
#define HD   64
#define Ee   8
#define DFF  4096
#define Tt   (Bz * Sq)     // 2048 tokens

static __device__ __forceinline__ float sig_(float x) { return 1.f / (1.f + __expf(-x)); }

static __device__ __forceinline__ float f4c(const float4& v, int u) {
  return u == 0 ? v.x : u == 1 ? v.y : u == 2 ? v.z : v.w;
}

// build a bf16x16 fragment from four float4s (element order preserved)
static __device__ __forceinline__ bf16x16 mk16(float4 a, float4 b, float4 c, float4 d) {
  bf16x16 r;
  r[0]  = (__bf16)a.x; r[1]  = (__bf16)a.y; r[2]  = (__bf16)a.z; r[3]  = (__bf16)a.w;
  r[4]  = (__bf16)b.x; r[5]  = (__bf16)b.y; r[6]  = (__bf16)b.z; r[7]  = (__bf16)b.w;
  r[8]  = (__bf16)c.x; r[9]  = (__bf16)c.y; r[10] = (__bf16)c.z; r[11] = (__bf16)c.w;
  r[12] = (__bf16)d.x; r[13] = (__bf16)d.y; r[14] = (__bf16)d.z; r[15] = (__bf16)d.w;
  return r;
}

// ---------------- generic bf16-WMMA GEMM: C = A@B (+bias), optional expert routing ----
// block: 256 thr = 8 waves; block tile 64(M)x128(N); wave tile 32x32 (2x2 frags); K step 32.
// LDS layouts keep 32 k contiguous per A row / B column (lane half h reads k = 16h+s) so that every
// fragment load is ONE contiguous 32-byte LDS read (ds_load_b128 pair).
//   A: As[row*32 + k]          (row-major: per-row K-runs contiguous)
//   B: Bs[n*32 + k]          (transposed: per-column K-runs contiguous)
__global__ __launch_bounds__(256)
void k_gemm(const float* __restrict__ A, int lda,
            const float* __restrict__ Bm, long long bStride, int ldb,
            float* __restrict__ C, int ldc,
            int M, int N, int K,
            const float* __restrict__ bias, int biasStride,
            const int* __restrict__ counts, const int* __restrict__ offsets,
            const int* __restrict__ rowmap)
{
  const int e     = blockIdx.z;
  const int Meff  = counts ? counts[e] : M;
  const int obase = offsets ? offsets[e] : 0;
  const int mt    = blockIdx.y * 64;
  if (mt >= Meff) return;
  const int n0 = blockIdx.x * 128;
  const float* Bp = Bm + (long long)e * bStride;
  const float* bp = bias ? (bias + (long long)e * biasStride) : nullptr;

  __shared__ __attribute__((aligned(32))) __bf16 As[64 * 32];
  __shared__ __attribute__((aligned(32))) __bf16 Bs[128 * 32];

  const int tid  = threadIdx.x;
  const int lane = tid & 31;
  const int wave = tid >> 5;
  const int wm   = (wave >> 2) * 32;   // 0,32
  const int wn   = (wave & 3) * 32;    // 0..96
  const int l15  = lane & 15;
  const int sel  = (lane & 16) ? 16 : 0;   // K-run selector for both A and B fragments

  f32x8 acc[2][2];
#pragma unroll
  for (int i = 0; i < 2; ++i)
#pragma unroll
    for (int j = 0; j < 2; ++j) acc[i][j] = {};

  for (int k0 = 0; k0 < K; k0 += 32) {
    if (k0 + 32 < K)   // hint next B panel into cache (global_prefetch_b8)
      __builtin_prefetch(Bp + (long long)(k0 + 32) * ldb + n0 + (tid & 127), 0, 1);

    // ---- stage A tile (64 rows x 32 k), float4 global loads -> bf16x4 LDS stores ----
    for (int i = tid; i < 512; i += 256) {
      int r = i >> 3, c4 = (i & 7) << 2;
      int m = mt + r;
      float4 v = make_float4(0.f, 0.f, 0.f, 0.f);
      if (m < Meff) {
        int slot = obase + m;
        int arow = rowmap ? rowmap[slot] : slot;
        v = *(const float4*)(A + (long long)arow * lda + k0 + c4);
      }
      bf16x4 hv;
      hv[0] = (__bf16)v.x; hv[1] = (__bf16)v.y; hv[2] = (__bf16)v.z; hv[3] = (__bf16)v.w;
      int p = c4;
      *(bf16x4*)(As + r * 32 + p) = hv;
    }
    // ---- stage B tile (32 k x 128 n) transposed, 2 rows x 4 cols per chunk ----
    for (int i = tid; i < 512; i += 256) {
      int rp = (i >> 5) << 1;          // even k row
      int c4 = (i & 31) << 2;          // col
      float4 v0 = make_float4(0.f, 0.f, 0.f, 0.f), v1 = v0;
      if (n0 + c4 < N) {
        v0 = *(const float4*)(Bp + (long long)(k0 + rp)     * ldb + n0 + c4);
        v1 = *(const float4*)(Bp + (long long)(k0 + rp + 1) * ldb + n0 + c4);
      }
#pragma unroll
      for (int u = 0; u < 4; ++u) {
        bf16x2 pr;
        pr[0] = (__bf16)f4c(v0, u);
        pr[1] = (__bf16)f4c(v1, u);
        *(bf16x2*)(Bs + (c4 + u) * 32 + rp) = pr;
      }
    }
    __syncthreads();

    bf16x16 a0 = *(const bf16x16*)(As + (wm      + l15) * 32 + sel);
    bf16x16 a1 = *(const bf16x16*)(As + (wm + 16 + l15) * 32 + sel);
    bf16x16 b0 = *(const bf16x16*)(Bs + (wn      + l15) * 32 + sel);
    bf16x16 b1 = *(const bf16x16*)(Bs + (wn + 16 + l15) * 32 + sel);
    acc[0][0] = WMMA_BF16(a0, b0, acc[0][0]);
    acc[0][1] = WMMA_BF16(a0, b1, acc[0][1]);
    acc[1][0] = WMMA_BF16(a1, b0, acc[1][0]);
    acc[1][1] = WMMA_BF16(a1, b1, acc[1][1]);
    __syncthreads();
  }

  // C/D layout: vgpr r, lanes0-15: (M=r, N=lane); lanes16-31: (M=r+8, N=lane-16)
  const int hi8 = (lane & 16) ? 8 : 0;
#pragma unroll
  for (int i = 0; i < 2; ++i)
#pragma unroll
    for (int j = 0; j < 2; ++j) {
      int nn = n0 + wn + j * 16 + l15;
      if (nn >= N) continue;
      float bv = bp ? bp[nn] : 0.f;
#pragma unroll
      for (int r = 0; r < 8; ++r) {
        int m = mt + wm + i * 16 + r + hi8;
        if (m < Meff) C[(long long)(obase + m) * ldc + nn] = acc[i][j][r] + bv;
      }
    }
}

// ---------------- per-(token,head) dynamic score scale: 2*sigmoid(q . scale_w) ----------
__global__ void k_qscale(const float* __restrict__ q, const float* __restrict__ sw,
                         float* __restrict__ qsc) {
  int idx = blockIdx.x * blockDim.x + threadIdx.x;  // t*H + h
  if (idx >= Tt * Hh) return;
  int t = idx >> 4, h = idx & 15;
  float a = 0.f;
#pragma unroll 4
  for (int d = 0; d < HD; ++d) a += q[(long long)t * Dm + h * HD + d] * sw[h * HD + d];
  qsc[idx] = 2.f * sig_(a);
}

// ---------------- attention: one wave per (b,h,16-query tile); scores live in LDS ------
__global__ __launch_bounds__(32)
void k_attn(const float* __restrict__ Q, const float* __restrict__ Kg,
            const float* __restrict__ V, const float* __restrict__ qsc,
            float* __restrict__ O) {
  __shared__ float sc[16 * 1024];                                  // 64 KB scores
  __shared__ __attribute__((aligned(32))) __bf16 Vs[64 * 32];      // 4 KB V chunk [d][key]
  const int qt = blockIdx.x, h = blockIdx.y, bz = blockIdx.z;
  const int lane = threadIdx.x;
  const int l15 = lane & 15;
  const int hi8 = (lane & 16) ? 8 : 0;
  const int kb  = (lane & 16) ? 8 : 0;    // unused (A fragment also uses sel)
  const int sel = (lane & 16) ? 16 : 0;   // B-fragment K run
  const int tb = bz * Sq, qb = qt * 16, hc = h * HD;

  float rscale[8];
#pragma unroll
  for (int r = 0; r < 8; ++r)
    rscale[r] = qsc[(tb + qb + r + hi8) * Hh + h] * 0.125f;  // 1/sqrt(64) * gate

  // scores = gate * (Q Kt)/sqrt(HD), WMMA tile by tile, float4 global reads
  for (int kt = 0; kt < Sq / 16; ++kt) {
    f32x8 c = {};
#pragma unroll
    for (int ch = 0; ch < 2; ++ch) {       // HD=64 -> two K=32 steps
      const float4* qp = (const float4*)(Q + (long long)(tb + qb + l15) * Dm + hc + ch * 32);
      int i0 = sel >> 2;                   // 0 or 4
      bf16x16 a = mk16(qp[i0], qp[i0 + 1], qp[i0 + 2], qp[i0 + 3]);
      const float4* kp =
          (const float4*)(Kg + (long long)(tb + kt * 16 + l15) * Dm + hc + ch * 32 + sel);
      bf16x16 b = mk16(kp[0], kp[1], kp[2], kp[3]);
      c = WMMA_BF16(a, b, c);
    }
#pragma unroll
    for (int r = 0; r < 8; ++r)
      sc[(r + hi8) * 1024 + kt * 16 + l15] = c[r] * rscale[r];
  }
  __syncthreads();

  // softmax over 1024 keys, one row at a time, whole wave cooperates
  for (int row = 0; row < 16; ++row) {
    float mx = -3.4e38f;
    for (int i = lane; i < 1024; i += 32) mx = fmaxf(mx, sc[row * 1024 + i]);
#pragma unroll
    for (int off = 16; off; off >>= 1) mx = fmaxf(mx, __shfl_xor(mx, off, 32));
    float sum = 0.f;
    for (int i = lane; i < 1024; i += 32) {
      float ev = __expf(sc[row * 1024 + i] - mx);
      sc[row * 1024 + i] = ev;
      sum += ev;
    }
#pragma unroll
    for (int off = 16; off; off >>= 1) sum += __shfl_xor(sum, off, 32);
    float inv = 1.f / sum;
    for (int i = lane; i < 1024; i += 32) sc[row * 1024 + i] *= inv;
  }
  __syncthreads();

  // out = P @ V  (16 x 64 accumulator = 4 fragments); V chunk staged transposed in LDS
  f32x8 o[4];
#pragma unroll
  for (int f = 0; f < 4; ++f) o[f] = {};
  for (int kc = 0; kc < Sq / 32; ++kc) {
    for (int i = lane; i < 512; i += 32) {      // 32 keys x 64 d, float4 quads
      int key = i & 31, d4 = (i >> 5) << 2;
      float4 v = *(const float4*)(V + (long long)(tb + kc * 32 + key) * Dm + hc + d4);
      Vs[(d4 + 0) * 32 + key] = (__bf16)v.x;
      Vs[(d4 + 1) * 32 + key] = (__bf16)v.y;
      Vs[(d4 + 2) * 32 + key] = (__bf16)v.z;
      Vs[(d4 + 3) * 32 + key] = (__bf16)v.w;
    }
    __syncthreads();
    const float4* sp = (const float4*)(sc + l15 * 1024 + kc * 32);
    int i0 = sel >> 2;
    bf16x16 a = mk16(sp[i0], sp[i0 + 1], sp[i0 + 2], sp[i0 + 3]);
#pragma unroll
    for (int f = 0; f < 4; ++f) {
      bf16x16 b = *(const bf16x16*)(Vs + (f * 16 + l15) * 32 + sel);
      o[f] = WMMA_BF16(a, b, o[f]);
    }
    __syncthreads();
  }
#pragma unroll
  for (int f = 0; f < 4; ++f) {
    int col = hc + f * 16 + l15;
#pragma unroll
    for (int r = 0; r < 8; ++r)
      O[(long long)(tb + qb + r + hi8) * Dm + col] = o[f][r];
  }
}

// ---------------- elementwise: og = o * sigmoid(gtmp) ----------------
__global__ void k_gate_mul(const float* __restrict__ o, const float* __restrict__ gt,
                           float* __restrict__ og, int n) {
  int i = blockIdx.x * blockDim.x + threadIdx.x;
  if (i < n) og[i] = o[i] * sig_(gt[i]);
}

// ---------------- block reduce helper ----------------
static __device__ __forceinline__ float blk_sum(float v, float* red) {
  int t = threadIdx.x;
  red[t] = v; __syncthreads();
  for (int st = 128; st > 0; st >>= 1) {
    if (t < st) red[t] += red[t + st];
    __syncthreads();
  }
  float r = red[0]; __syncthreads();
  return r;
}

// ---------------- LN(a (+b)) * g + beta ----------------
__global__ __launch_bounds__(256)
void k_ln(const float* __restrict__ a, const float* __restrict__ b,
          const float* __restrict__ g, const float* __restrict__ be,
          float* __restrict__ out, int Dlen) {
  __shared__ float red[256];
  long long row = blockIdx.x;
  float s = 0.f, s2 = 0.f;
  for (int i = threadIdx.x; i < Dlen; i += 256) {
    float v = a[row * Dlen + i];
    if (b) v += b[row * Dlen + i];
    s += v; s2 += v * v;
  }
  float mean = blk_sum(s, red) / Dlen;
  float var  = blk_sum(s2, red) / Dlen - mean * mean;
  float inv  = rsqrtf(var + 1e-5f);
  for (int i = threadIdx.x; i < Dlen; i += 256) {
    float v = a[row * Dlen + i];
    if (b) v += b[row * Dlen + i];
    out[row * Dlen + i] = (v - mean) * inv * g[i] + be[i];
  }
}

// ---------------- per-expert LN + exact GELU, in place on h (rows are slots) ----------
__global__ __launch_bounds__(256)
void k_lngelu(float* __restrict__ h, const float* __restrict__ ln_g,
              const float* __restrict__ ln_b, const int* __restrict__ slot2exp) {
  __shared__ float red[256];
  long long row = blockIdx.x;
  int e = slot2exp[row];
  const float* g  = ln_g + (long long)e * DFF;
  const float* be = ln_b + (long long)e * DFF;
  float s = 0.f, s2 = 0.f;
  for (int i = threadIdx.x; i < DFF; i += 256) {
    float v = h[row * DFF + i];
    s += v; s2 += v * v;
  }
  float mean = blk_sum(s, red) / DFF;
  float var  = blk_sum(s2, red) / DFF - mean * mean;
  float inv  = rsqrtf(var + 1e-5f);
  for (int i = threadIdx.x; i < DFF; i += 256) {
    float n = (h[row * DFF + i] - mean) * inv * g[i] + be[i];
    h[row * DFF + i] = 0.5f * n * (1.f + erff(n * 0.70710678118654752f));
  }
}

// ---------------- MoE gating: logits, top-2, e_sel = max index; count tokens ----------
__global__ void k_moe_gate(const float* __restrict__ x, const float* __restrict__ mgw,
                           const float* __restrict__ mgb, int* __restrict__ sel,
                           int* __restrict__ counts) {
  int t = blockIdx.x * blockDim.x + threadIdx.x;
  if (t >= Tt) return;
  float acc[Ee];
#pragma unroll
  for (int e = 0; e < Ee; ++e) acc[e] = mgb[e];
  for (int d = 0; d < Dm; ++d) {
    float xv = x[(long long)t * Dm + d];
#pragma unroll
    for (int e = 0; e < Ee; ++e) acc[e] += xv * mgw[d * Ee + e];
  }
  float v1 = -3.4e38f, v2 = -3.4e38f; int i1 = 0, i2 = 0;
#pragma unroll
  for (int e = 0; e < Ee; ++e) {
    float l = acc[e];
    if (l > v1) { v2 = v1; i2 = i1; v1 = l; i1 = e; }
    else if (l > v2) { v2 = l; i2 = e; }
  }
  int es = (i1 > i2) ? i1 : i2;     // reference: larger selected index wins; wsum == 1
  sel[t] = es;
  atomicAdd(&counts[es], 1);
}

__global__ void k_init(int* counts, int* cursor) {
  if (threadIdx.x < Ee) { counts[threadIdx.x] = 0; cursor[threadIdx.x] = 0; }
}
__global__ void k_scan(const int* counts, int* offsets) {
  if (threadIdx.x == 0) {
    int a = 0;
    for (int e = 0; e < Ee; ++e) { offsets[e] = a; a += counts[e]; }
  }
}
__global__ void k_scatter(const int* __restrict__ sel, const int* __restrict__ offsets,
                          int* __restrict__ cursor, int* __restrict__ slot2tok,
                          int* __restrict__ tok2slot, int* __restrict__ slot2exp) {
  int t = blockIdx.x * blockDim.x + threadIdx.x;
  if (t >= Tt) return;
  int e = sel[t];
  int p = atomicAdd(&cursor[e], 1);
  int slot = offsets[e] + p;
  slot2tok[slot] = t; tok2slot[t] = slot; slot2exp[slot] = e;
}

// ---------------- final: out = LN2(2*x + res_scale[e]*y2) ----------------
__global__ __launch_bounds__(256)
void k_final(const float* __restrict__ x, const float* __restrict__ y2,
             const int* __restrict__ tok2slot, const int* __restrict__ sel,
             const float* __restrict__ res, const float* __restrict__ g,
             const float* __restrict__ be, float* __restrict__ out) {
  __shared__ float red[256];
  long long t = blockIdx.x;
  long long slot = tok2slot[t];
  float rs = res[sel[t]];
  float s = 0.f, s2 = 0.f;
  for (int i = threadIdx.x; i < Dm; i += 256) {
    float v = 2.f * x[t * Dm + i] + rs * y2[slot * Dm + i];
    s += v; s2 += v * v;
  }
  float mean = blk_sum(s, red) / Dm;
  float var  = blk_sum(s2, red) / Dm - mean * mean;
  float inv  = rsqrtf(var + 1e-5f);
  for (int i = threadIdx.x; i < Dm; i += 256) {
    float v = 2.f * x[t * Dm + i] + rs * y2[slot * Dm + i];
    out[t * Dm + i] = (v - mean) * inv * g[i] + be[i];
  }
}

// =====================================================================================
extern "C" void kernel_launch(void* const* d_in, const int* in_sizes, int n_in,
                              void* d_out, int out_size, void* d_ws, size_t ws_size,
                              hipStream_t stream) {
  (void)in_sizes; (void)n_in; (void)out_size; (void)ws_size;
  const float* src   = (const float*)d_in[0];
  const float* qw    = (const float*)d_in[1];
  const float* kw    = (const float*)d_in[2];
  const float* vw    = (const float*)d_in[3];
  const float* ow    = (const float*)d_in[4];
  const float* gw    = (const float*)d_in[5];
  const float* gb    = (const float*)d_in[6];
  const float* sw    = (const float*)d_in[7];
  const float* n1g   = (const float*)d_in[8];
  const float* n1b   = (const float*)d_in[9];
  const float* n2g   = (const float*)d_in[10];
  const float* n2b   = (const float*)d_in[11];
  const float* mgw   = (const float*)d_in[12];
  const float* mgb   = (const float*)d_in[13];
  const float* w1    = (const float*)d_in[14];
  const float* b1    = (const float*)d_in[15];
  const float* lng   = (const float*)d_in[16];
  const float* lnb   = (const float*)d_in[17];
  const float* w2    = (const float*)d_in[18];
  const float* b2    = (const float*)d_in[19];
  const float* resc  = (const float*)d_in[20];

  const long long TD = (long long)Tt * Dm;        // 2M floats per activation
  float* f  = (float*)d_ws;
  float* q    = f + 0 * TD;
  float* kx   = f + 1 * TD;
  float* v    = f + 2 * TD;
  float* o    = f + 3 * TD;
  float* gtmp = f + 4 * TD;
  float* og   = f + 5 * TD;
  float* ao   = f + 6 * TD;
  float* x    = f + 7 * TD;
  float* y2   = f + 8 * TD;
  float* h    = f + 9 * TD;                        // Tt * DFF = 4*TD floats
  float* qsc  = f + 13 * TD;                       // Tt*Hh floats
  int*   ip        = (int*)(qsc + (long long)Tt * Hh);
  int*   sel       = ip;            // Tt
  int*   counts    = ip + Tt;       // Ee
  int*   offsets   = counts + Ee;   // Ee
  int*   cursor    = offsets + Ee;  // Ee
  int*   slot2tok  = cursor + Ee;   // Tt
  int*   tok2slot  = slot2tok + Tt; // Tt
  int*   slot2exp  = tok2slot + Tt; // Tt

  // ---- QKV projections (bf16 WMMA, fp32 accumulate) ----
  dim3 gP(Dm / 128, Tt / 64, 1);
  k_gemm<<<gP, 256, 0, stream>>>(src, Dm, qw, 0, Dm, q,  Dm, Tt, Dm, Dm, nullptr, 0, nullptr, nullptr, nullptr);
  k_gemm<<<gP, 256, 0, stream>>>(src, Dm, kw, 0, Dm, kx, Dm, Tt, Dm, Dm, nullptr, 0, nullptr, nullptr, nullptr);
  k_gemm<<<gP, 256, 0, stream>>>(src, Dm, vw, 0, Dm, v,  Dm, Tt, Dm, Dm, nullptr, 0, nullptr, nullptr, nullptr);

  // ---- per-(token,head) score scale ----
  k_qscale<<<(Tt * Hh + 255) / 256, 256, 0, stream>>>(q, sw, qsc);

  // ---- attention ----
  k_attn<<<dim3(Sq / 16, Hh, Bz), 32, 0, stream>>>(q, kx, v, qsc, o);

  // ---- output gate: sigmoid(o@gate_w + gb) * o, then @ow ----
  k_gemm<<<gP, 256, 0, stream>>>(o, Dm, gw, 0, Dm, gtmp, Dm, Tt, Dm, Dm, gb, 0, nullptr, nullptr, nullptr);
  k_gate_mul<<<(int)(TD / 256), 256, 0, stream>>>(o, gtmp, og, (int)TD);
  k_gemm<<<gP, 256, 0, stream>>>(og, Dm, ow, 0, Dm, ao, Dm, Tt, Dm, Dm, nullptr, 0, nullptr, nullptr, nullptr);

  // ---- LN1(src + attn_out) ----
  k_ln<<<Tt, 256, 0, stream>>>(src, ao, n1g, n1b, x, Dm);

  // ---- MoE routing (top-1-by-max-index, weight == 1) ----
  k_init<<<1, 32, 0, stream>>>(counts, cursor);
  k_moe_gate<<<Tt / 256, 256, 0, stream>>>(x, mgw, mgb, sel, counts);
  k_scan<<<1, 32, 0, stream>>>(counts, offsets);
  k_scatter<<<Tt / 256, 256, 0, stream>>>(sel, offsets, cursor, slot2tok, tok2slot, slot2exp);

  // ---- expert GEMM1: h = gather(x) @ w1[e] + b1[e]  (ragged via counts/offsets) ----
  k_gemm<<<dim3(DFF / 128, Tt / 64, Ee), 256, 0, stream>>>(
      x, Dm, w1, (long long)Dm * DFF, DFF, h, DFF, Tt, DFF, Dm,
      b1, DFF, counts, offsets, slot2tok);

  // ---- per-expert LN + exact GELU over DFF ----
  k_lngelu<<<Tt, 256, 0, stream>>>(h, lng, lnb, slot2exp);

  // ---- expert GEMM2: y2 = h @ w2[e] + b2[e] ----
  k_gemm<<<dim3(Dm / 128, Tt / 64, Ee), 256, 0, stream>>>(
      h, DFF, w2, (long long)DFF * Dm, Dm, y2, Dm, Tt, Dm, DFF,
      b2, Dm, counts, offsets, nullptr);

  // ---- final LN2(2x + res*y2) -> out ----
  k_final<<<Tt, 256, 0, stream>>>(x, y2, tok2slot, sel, resc, n2g, n2b, (float*)d_out);
}
